// BiLSTMWithAttention_38147899523467
// MI455X (gfx1250) — matrix operand order repaired and checked by reading, not compiled, not run on hardware
//
#include <hip/hip_runtime.h>
#include <cstdint>
#include <cstddef>

// Problem dims (fixed by the reference)
#define B_ 64
#define C_ 64
#define T_ 2048
#define H_ 128   // 2H = 256, 4H = 512

typedef __attribute__((ext_vector_type(16))) _Float16     v16h;
typedef __attribute__((ext_vector_type(8)))  _Float16     v8h;
typedef __attribute__((ext_vector_type(8)))  float        v8f;
typedef __attribute__((ext_vector_type(4)))  unsigned int v4u;

// WMMA A/B operand assembled from two 16-byte loads
union AF  { v16h v;  v4u q[2]; };
// 8 packed f16 (one 16-byte load/store)
union AF8 { v8h  h8; v4u q;    };

__device__ __forceinline__ float sigm(float x)      { return 1.0f / (1.0f + __expf(-x)); }
__device__ __forceinline__ float tanh_fast(float x) { float e = __expf(2.0f * x); return 1.0f - 2.0f / (e + 1.0f); }

// ---------------------------------------------------------------------------
// Pack kernels: convert f32 inputs into WMMA-friendly f16 layouts in d_ws.
// ---------------------------------------------------------------------------

// x (B,C,T) f32  ->  xT (B,T,C) f16  (row-major rows of length C=64)
__global__ void pack_x(const float* __restrict__ x, _Float16* __restrict__ xT) {
    int tid = blockIdx.x * blockDim.x + threadIdx.x;
    if (tid >= B_ * T_ * C_) return;
    int c = tid % C_;
    int t = (tid / C_) % T_;
    int b = tid / (C_ * T_);
    xT[tid] = (_Float16)x[(size_t)b * C_ * T_ + (size_t)c * T_ + t];
}

// Pack weight W[Ntot][Ktot] (row-major) into B-operand fragments for
// v_wmma_f32_16x16x32_f16.  B layout (CDNA5 ISA, 16-bit B 32x16):
//   lane l holds column N = l&15; element e holds K = (e&7) + 8*(l>>4) + 16*(e>>3).
// Fragment order: ft = kt*(Ntot/16) + nt; each fragment is [lane][e] = 512 f16.
__global__ void pack_w(const float* __restrict__ W, _Float16* __restrict__ dst,
                       int Ntot, int Ktot) {
    int total = (Ktot / 32) * (Ntot / 16) * 512;
    int tid = blockIdx.x * blockDim.x + threadIdx.x;
    if (tid >= total) return;
    int e    = tid & 15;
    int lane = (tid >> 4) & 31;
    int ft   = tid >> 9;
    int nt16 = Ntot / 16;
    int kt = ft / nt16, nt = ft % nt16;
    int n = nt * 16 + (lane & 15);
    int k = kt * 32 + (e & 7) + 8 * (lane >> 4) + 16 * (e >> 3);
    dst[tid] = (_Float16)W[(size_t)n * Ktot + k];
}

// bsum[dir][g] = bih[g] + bhh[g]
__global__ void pack_bias(const float* __restrict__ bihf, const float* __restrict__ bhhf,
                          const float* __restrict__ bihb, const float* __restrict__ bhhb,
                          float* __restrict__ bsum) {
    int tid = blockIdx.x * blockDim.x + threadIdx.x;
    if (tid >= 2 * 4 * H_) return;
    int g = tid & (4 * H_ - 1);
    bsum[tid] = (tid < 4 * H_) ? (bihf[g] + bhhf[g]) : (bihb[g] + bhhb[g]);
}

// ---------------------------------------------------------------------------
// Time-parallel input projection: zx[dir][t][jg][b] = x_t . Wih^T + bih + bhh.
// One wave per (dir, 16-batch tile, t); 32 N-tiles x 2 K-chunks = 64 WMMAs.
// Output layout puts 8 consecutive batch elements per lane contiguously so the
// recurrent kernel's accumulator init is a single b128 load per gate.
// ---------------------------------------------------------------------------
__global__ __launch_bounds__(256) void zx_gemm(
    const _Float16* __restrict__ xT,    // [B][T][C]
    const _Float16* __restrict__ WihP,  // [2][2*32 frags][512]
    const float*    __restrict__ bsum,  // [2][4H]
    _Float16*       __restrict__ zx)    // [2][T][4H][B]
{
    const int lane = threadIdx.x & 31;
    const int wv   = threadIdx.x >> 5;
    const int gw   = blockIdx.x * 8 + wv;       // 0 .. 2*4*2048-1
    const int t     = gw & (T_ - 1);
    const int btile = (gw >> 11) & 3;
    const int dir   = gw >> 13;
    const int b0 = btile * 16;
    const int row = lane & 15, half = lane >> 4, koff = 8 * half;

    // A fragments: 16 batch rows at fixed t (K = C = 64)
    AF Ax[2];
    const _Float16* xrow = xT + ((size_t)(b0 + row) * T_ + t) * C_;
#pragma unroll
    for (int ch = 0; ch < 2; ++ch) {
        Ax[ch].q[0] = *(const v4u*)(xrow + ch * 32 + koff);
        Ax[ch].q[1] = *(const v4u*)(xrow + ch * 32 + koff + 16);
    }

    const _Float16* wih = WihP + (size_t)dir * (2 * 32 * 512);
    const size_t zbase = ((size_t)dir * T_ + t) * (4 * H_);

    for (int nt = 0; nt < 32; ++nt) {
        const float bb = bsum[dir * 512 + nt * 16 + (lane & 15)];
        v8f acc;
#pragma unroll
        for (int v = 0; v < 8; ++v) acc[v] = bb;
#pragma unroll
        for (int kt = 0; kt < 2; ++kt) {
            AF f; const _Float16* p = wih + ((size_t)(kt * 32 + nt)) * 512 + lane * 16;
            f.q[0] = *(const v4u*)p; f.q[1] = *(const v4u*)(p + 8);
            acc = __builtin_amdgcn_wmma_f32_16x16x32_f16(false, Ax[kt].v, false, f.v, (short)0, acc, false, false);
        }
        AF8 o;
#pragma unroll
        for (int v = 0; v < 8; ++v) o.h8[v] = (_Float16)acc[v];
        *(v4u*)(zx + (zbase + nt * 16 + (lane & 15)) * B_ + b0 + 8 * half) = o.q;
    }
}

// ---------------------------------------------------------------------------
// Bidirectional LSTM recurrence.
// grid = 8 blocks: dir(2) x batch-tile(4, 16 rows each).  8 waves/block, each
// wave owns 16 hidden columns and computes i/f/g/o for them.
//   UZX=1: z_x precomputed -> 16 WMMAs/step, <256 VGPRs (no vgpr-msb churn)
//   UZX=0: x-part in-loop  -> 24 WMMAs/step (fallback when ws is small)
// h is LDS-resident f16; c stays in f32 registers for all 2048 steps.
// ---------------------------------------------------------------------------
template <int UZX>
__global__ __launch_bounds__(256) void lstm_rec(
    const _Float16* __restrict__ zx,     // [2][T][4H][B]  (UZX=1)
    const _Float16* __restrict__ xT,     // [B][T][C]      (UZX=0)
    const _Float16* __restrict__ WihP,   // [2][2*32 frags][512]
    const _Float16* __restrict__ WhhP,   // [2][4*32 frags][512]
    const float*    __restrict__ bsum,   // [2][4H]
    _Float16*       __restrict__ lstm16) // [B][T][2H]
{
    const int dir  = blockIdx.x >> 2;
    const int b0   = (blockIdx.x & 3) * 16;
    const int lane = threadIdx.x & 31;
    const int wv   = threadIdx.x >> 5;      // wave 0..7 -> hidden cols [16wv,16wv+16)
    const int row  = lane & 15;             // A-matrix row (batch)
    const int half = lane >> 4;
    const int koff = 8 * half;              // A-fragment K base (ISA 16-bit A layout)

    __shared__ _Float16 hbuf[16][136];      // padded stride: 272B = 17*16B

    for (int i = threadIdx.x; i < 16 * 136; i += 256)
        ((_Float16*)hbuf)[i] = (_Float16)0.0f;

    // Register-resident B fragments (constant across all 2048 steps)
    v16h Bx[4][2];   // Wih: only used when UZX==0
    v16h Bh[4][4];   // Whh: 4 gates x 4 K-chunks (K=128)
    const _Float16* whh = WhhP + (size_t)dir * (4 * 32 * 512);
#pragma unroll
    for (int g = 0; g < 4; ++g) {
        const int nt = g * 8 + wv;          // N-tile within 4H
#pragma unroll
        for (int kt = 0; kt < 4; ++kt) {
            AF f; const _Float16* p = whh + ((size_t)(kt * 32 + nt)) * 512 + lane * 16;
            f.q[0] = *(const v4u*)p; f.q[1] = *(const v4u*)(p + 8);
            Bh[g][kt] = f.v;
        }
    }
    if (UZX == 0) {
        const _Float16* wih = WihP + (size_t)dir * (2 * 32 * 512);
#pragma unroll
        for (int g = 0; g < 4; ++g) {
            const int nt = g * 8 + wv;
#pragma unroll
            for (int kt = 0; kt < 2; ++kt) {
                AF f; const _Float16* p = wih + ((size_t)(kt * 32 + nt)) * 512 + lane * 16;
                f.q[0] = *(const v4u*)p; f.q[1] = *(const v4u*)(p + 8);
                Bx[g][kt] = f.v;
            }
        }
    }

    float bias[4];
#pragma unroll
    for (int g = 0; g < 4; ++g)
        bias[g] = (UZX == 0) ? bsum[dir * 512 + g * 128 + wv * 16 + (lane & 15)] : 0.0f;

    v8f cst;                                // cell state (f32, in registers)
#pragma unroll
    for (int v = 0; v < 8; ++v) cst[v] = 0.0f;

    const size_t zcol = (size_t)wv * 16 + (lane & 15);   // column within a gate
    __syncthreads();

    for (int ts = 0; ts < T_; ++ts) {
        const int t = dir ? (T_ - 1 - ts) : ts;

        v8f acc[4];
        AF Ax[2];
        if (UZX) {
            // Accumulator init from precomputed z_x (includes biases):
            // one b128 load per gate, 8 consecutive batch elements per lane.
            const size_t zt = ((size_t)dir * T_ + t) * (4 * H_);
#pragma unroll
            for (int g = 0; g < 4; ++g) {
                AF8 zv;
                zv.q = *(const v4u*)(zx + (zt + (size_t)g * H_ + zcol) * B_ + b0 + 8 * half);
#pragma unroll
                for (int v = 0; v < 8; ++v) acc[g][v] = (float)zv.h8[v];
            }
            if (ts + 1 < T_) {
                const int tn = dir ? (t - 1) : (t + 1);
                const size_t ztn = ((size_t)dir * T_ + tn) * (4 * H_);
#pragma unroll
                for (int g = 0; g < 4; ++g)
                    __builtin_prefetch(zx + (ztn + (size_t)g * H_ + zcol) * B_ + b0 + 8 * half, 0, 3);
            }
        } else {
            // A fragments for x-part (16 batch rows x C=64), global f16
            const _Float16* xrow = xT + ((size_t)(b0 + row) * T_ + t) * C_;
#pragma unroll
            for (int ch = 0; ch < 2; ++ch) {
                Ax[ch].q[0] = *(const v4u*)(xrow + ch * 32 + koff);
                Ax[ch].q[1] = *(const v4u*)(xrow + ch * 32 + koff + 16);
            }
            if (ts + 1 < T_) {
                const int tn = dir ? (t - 1) : (t + 1);
                __builtin_prefetch(xT + ((size_t)(b0 + row) * T_ + tn) * C_, 0, 3);
            }
#pragma unroll
            for (int g = 0; g < 4; ++g)
#pragma unroll
                for (int v = 0; v < 8; ++v) acc[g][v] = bias[g];
        }

        // A fragments for h-part (16 batch rows x H=128), from LDS
        AF Ah[4];
#pragma unroll
        for (int ch = 0; ch < 4; ++ch) {
            Ah[ch].q[0] = *(const v4u*)&hbuf[row][ch * 32 + koff];
            Ah[ch].q[1] = *(const v4u*)&hbuf[row][ch * 32 + koff + 16];
        }
        __syncthreads();  // all waves have read h before anyone overwrites it

#pragma unroll
        for (int g = 0; g < 4; ++g) {
            if (UZX == 0) {
                acc[g] = __builtin_amdgcn_wmma_f32_16x16x32_f16(false, Ax[0].v, false, Bx[g][0], (short)0, acc[g], false, false);
                acc[g] = __builtin_amdgcn_wmma_f32_16x16x32_f16(false, Ax[1].v, false, Bx[g][1], (short)0, acc[g], false, false);
            }
#pragma unroll
            for (int kc = 0; kc < 4; ++kc)
                acc[g] = __builtin_amdgcn_wmma_f32_16x16x32_f16(false, Ah[kc].v, false, Bh[g][kc], (short)0, acc[g], false, false);
        }

        // Gates + state update.  D layout: VGPR v, lane l -> M = v + 8*(l>>4), N = l&15.
        const int j = wv * 16 + (lane & 15);
#pragma unroll
        for (int v = 0; v < 8; ++v) {
            float iv = sigm(acc[0][v]);
            float fv = sigm(acc[1][v]);
            float gv = tanh_fast(acc[2][v]);
            float ov = sigm(acc[3][v]);
            float cv = fv * cst[v] + iv * gv;
            cst[v] = cv;
            float hv = ov * tanh_fast(cv);
            int m = v + 8 * half;
            hbuf[m][j] = (_Float16)hv;
            lstm16[((size_t)(b0 + m) * T_ + t) * (2 * H_) + dir * H_ + j] = (_Float16)hv;
        }
        __syncthreads();
    }
}

// ---------------------------------------------------------------------------
// Attention scores: score[r] = Wu . tanh(Wa . lstm[r] + ba) + bu, r = b*T+t.
// One wave per 16-row tile; full N=256 swept in 16 tiles of 8 WMMAs (K=256).
// ---------------------------------------------------------------------------
__global__ __launch_bounds__(256) void attn_scores(
    const _Float16* __restrict__ lstm16, // [B*T][256]
    const _Float16* __restrict__ WaP,    // [8*16 frags][512]
    const float*    __restrict__ ba,     // [256]
    const float*    __restrict__ Wu,     // [256]
    const float*    __restrict__ bu,     // [1]
    float*          __restrict__ scores) // [B*T]
{
    const int lane = threadIdx.x & 31;
    const int wv   = threadIdx.x >> 5;
    const int r0   = (blockIdx.x * 8 + wv) * 16;
    const int row  = lane & 15, half = lane >> 4, koff = 8 * half;

    const _Float16* arow = lstm16 + (size_t)(r0 + row) * 256;
    AF Aa[8];
#pragma unroll
    for (int ch = 0; ch < 8; ++ch) {
        Aa[ch].q[0] = *(const v4u*)(arow + ch * 32 + koff);
        Aa[ch].q[1] = *(const v4u*)(arow + ch * 32 + koff + 16);
    }

    v8f sacc;
#pragma unroll
    for (int v = 0; v < 8; ++v) sacc[v] = 0.0f;

    for (int nt = 0; nt < 16; ++nt) {
        const float bb = ba[nt * 16 + (lane & 15)];
        const float wu = Wu[nt * 16 + (lane & 15)];
        v8f acc;
#pragma unroll
        for (int v = 0; v < 8; ++v) acc[v] = bb;
#pragma unroll
        for (int kt = 0; kt < 8; ++kt) {
            AF f; const _Float16* p = WaP + ((size_t)(kt * 16 + nt)) * 512 + lane * 16;
            f.q[0] = *(const v4u*)p; f.q[1] = *(const v4u*)(p + 8);
            acc = __builtin_amdgcn_wmma_f32_16x16x32_f16(false, Aa[kt].v, false, f.v, (short)0, acc, false, false);
        }
#pragma unroll
        for (int v = 0; v < 8; ++v) sacc[v] += wu * tanh_fast(acc[v]);
    }

    // Reduce over the 16 columns held by each 16-lane half (rows split by half).
#pragma unroll
    for (int v = 0; v < 8; ++v) {
        float val = sacc[v];
        val += __shfl_xor(val, 1, 32);
        val += __shfl_xor(val, 2, 32);
        val += __shfl_xor(val, 4, 32);
        val += __shfl_xor(val, 8, 32);
        sacc[v] = val;
    }
    if ((lane & 15) == 0) {
#pragma unroll
        for (int v = 0; v < 8; ++v)
            scores[r0 + v + 8 * half] = sacc[v] + bu[0];
    }
}

// ---------------------------------------------------------------------------
// Per-batch softmax over T and weighted sum: out[b,h] = sum_t w[b,t]*lstm[b,t,h]
// ---------------------------------------------------------------------------
__global__ __launch_bounds__(256) void attn_finish(
    const _Float16* __restrict__ lstm16,
    const float*    __restrict__ scores,
    float*          __restrict__ out)    // [B][256]
{
    const int b = blockIdx.x;
    const int tid = threadIdx.x;
    __shared__ float wbuf[T_];
    __shared__ float red[256];

    float m = -1e30f;
#pragma unroll
    for (int k = 0; k < T_ / 256; ++k)
        m = fmaxf(m, scores[(size_t)b * T_ + tid + k * 256]);
    red[tid] = m; __syncthreads();
    for (int s = 128; s > 0; s >>= 1) { if (tid < s) red[tid] = fmaxf(red[tid], red[tid + s]); __syncthreads(); }
    const float mx = red[0];
    __syncthreads();

    float ssum = 0.0f;
#pragma unroll
    for (int k = 0; k < T_ / 256; ++k) {
        float e = __expf(scores[(size_t)b * T_ + tid + k * 256] - mx);
        wbuf[tid + k * 256] = e;
        ssum += e;
    }
    red[tid] = ssum; __syncthreads();
    for (int s = 128; s > 0; s >>= 1) { if (tid < s) red[tid] += red[tid + s]; __syncthreads(); }
    const float Z = red[0];
    __syncthreads();

    const _Float16* base = lstm16 + (size_t)b * T_ * 256 + tid;
    float acc = 0.0f;
    for (int t = 0; t < T_; ++t)
        acc += wbuf[t] * (float)base[(size_t)t * 256];
    out[b * 256 + tid] = acc / Z;
}

// ---------------------------------------------------------------------------
extern "C" void kernel_launch(void* const* d_in, const int* in_sizes, int n_in,
                              void* d_out, int out_size, void* d_ws, size_t ws_size,
                              hipStream_t stream) {
    const float* x     = (const float*)d_in[0];
    const float* Wih_f = (const float*)d_in[1];
    const float* Whh_f = (const float*)d_in[2];
    const float* bih_f = (const float*)d_in[3];
    const float* bhh_f = (const float*)d_in[4];
    const float* Wih_b = (const float*)d_in[5];
    const float* Whh_b = (const float*)d_in[6];
    const float* bih_b = (const float*)d_in[7];
    const float* bhh_b = (const float*)d_in[8];
    const float* Wa    = (const float*)d_in[9];
    const float* ba    = (const float*)d_in[10];
    const float* Wu    = (const float*)d_in[11];
    const float* bu    = (const float*)d_in[12];
    float* out = (float*)d_out;

    // Workspace layout: ~85 MB base, +256 MB for the z_x path if it fits.
    char* ws = (char*)d_ws;
    size_t off = 0;
    auto take = [&](size_t bytes) { char* p = ws + off; off += (bytes + 255) & ~(size_t)255; return p; };
    _Float16* WihP   = (_Float16*)take((size_t)2 * 64  * 512 * sizeof(_Float16)); // 2 dirs x 64 frags
    _Float16* WhhP   = (_Float16*)take((size_t)2 * 128 * 512 * sizeof(_Float16)); // 2 dirs x 128 frags
    _Float16* WaP    = (_Float16*)take((size_t)128 * 512 * sizeof(_Float16));     // 128 frags
    float*    bsum   = (float*)   take((size_t)2 * 4 * H_ * sizeof(float));
    _Float16* xT16   = (_Float16*)take((size_t)B_ * T_ * C_ * sizeof(_Float16));      // 16 MB
    _Float16* lstm16 = (_Float16*)take((size_t)B_ * T_ * 2 * H_ * sizeof(_Float16)); // 64 MB
    float*    scores = (float*)   take((size_t)B_ * T_ * sizeof(float));             // 512 KB
    _Float16* zx     = (_Float16*)take((size_t)2 * T_ * 4 * H_ * B_ * sizeof(_Float16)); // 256 MB
    const bool use_zx = (off <= ws_size);   // ws_size is fixed per run -> deterministic
    (void)in_sizes; (void)n_in; (void)out_size;

    pack_x<<<(B_ * T_ * C_ + 255) / 256, 256, 0, stream>>>(x, xT16);
    pack_w<<<(2 * 32 * 512 + 255) / 256, 256, 0, stream>>>(Wih_f, WihP,                 512, 64);
    pack_w<<<(2 * 32 * 512 + 255) / 256, 256, 0, stream>>>(Wih_b, WihP + 64 * 512,      512, 64);
    pack_w<<<(4 * 32 * 512 + 255) / 256, 256, 0, stream>>>(Whh_f, WhhP,                 512, 128);
    pack_w<<<(4 * 32 * 512 + 255) / 256, 256, 0, stream>>>(Whh_b, WhhP + 128 * 512,     512, 128);
    pack_w<<<(8 * 16 * 512 + 255) / 256, 256, 0, stream>>>(Wa,    WaP,                  256, 256);
    pack_bias<<<(2 * 4 * H_ + 255) / 256, 256, 0, stream>>>(bih_f, bhh_f, bih_b, bhh_b, bsum);

    if (use_zx) {
        zx_gemm<<<2 * 4 * T_ / 8, 256, 0, stream>>>(xT16, WihP, bsum, zx);
        lstm_rec<1><<<8, 256, 0, stream>>>(zx, xT16, WihP, WhhP, bsum, lstm16);
    } else {
        lstm_rec<0><<<8, 256, 0, stream>>>(xT16 /*unused*/, xT16, WihP, WhhP, bsum, lstm16);
    }
    attn_scores<<<1024, 256, 0, stream>>>(lstm16, WaP, ba, Wu, bu, scores);
    attn_finish<<<64, 256, 0, stream>>>(lstm16, scores, out);
}
